// DMN_9844065042540
// MI455X (gfx1250) — matrix operand order repaired and checked by reading, not compiled, not run on hardware
//
#include <hip/hip_runtime.h>
#include <hip/hip_bf16.h>
#include <math.h>

// ---------------------------------------------------------------------------
// DMN forward for MI455X (gfx1250, wave32).
// All dense layers run through V_WMMA_F32_16X16X32_F16 (f16 in, f32 accum).
// ---------------------------------------------------------------------------

typedef __attribute__((ext_vector_type(16))) _Float16 v16h;
typedef __attribute__((ext_vector_type(8)))  _Float16 v8h;
typedef __attribute__((ext_vector_type(8)))  float    v8f;

#define HD   256          // embed = hidden = 256
#define VD   32000        // vocab
#define BD   64           // batch
#define NFD  20           // n facts
#define FLD  32           // fact len
#define QLD  16           // question len
#define ALEN 8            // answer length (reference alen)
#define G3H  768          // 3*H
#define SEQB 1            // SEQBEGIN

// ---------------- elementwise helpers ----------------

__global__ void k_f32_to_f16(const float* __restrict__ s, _Float16* __restrict__ d, int n) {
  int i = blockIdx.x * blockDim.x + threadIdx.x;
  if (i < n) d[i] = (_Float16)s[i];
}

__global__ void k_fill_f32(float* __restrict__ d, float v, int n) {
  int i = blockIdx.x * blockDim.x + threadIdx.x;
  if (i < n) d[i] = v;
}

__global__ void k_copy_f32(const float* __restrict__ s, float* __restrict__ d, int n) {
  int i = blockIdx.x * blockDim.x + threadIdx.x;
  if (i < n) d[i] = s[i];
}

// real_len-1 per row, clipped to [0, T-1]
__global__ void k_last_idx(const int* __restrict__ mask, int rows, int T, int* __restrict__ out) {
  int r = blockIdx.x * blockDim.x + threadIdx.x;
  if (r >= rows) return;
  int c = 0;
  for (int t = 0; t < T; ++t) c += (mask[r * T + t] == 0) ? 1 : 0;
  int idx = c - 1;
  idx = idx < 0 ? 0 : (idx > T - 1 ? T - 1 : idx);
  out[r] = idx;
}

// gather embedding row for token at position t of each sequence, -> f16
__global__ void k_embed_f16(const int* __restrict__ tok, int tokStride, int t,
                            const float* __restrict__ emb,
                            _Float16* __restrict__ dst, int rows) {
  int i = blockIdx.x * blockDim.x + threadIdx.x;
  if (i >= rows * HD) return;
  int r = i / HD, j = i % HD;
  int tk = tok[r * tokStride + t];
  dst[i] = (_Float16)emb[(size_t)tk * HD + j];
}

// ---------------- WMMA GEMM:  C[M,N] = act(A[M,K] * W[N,K]^T + bias) ----------------
// grid = (M/16, N/128); block = 256 (8 waves, each wave owns a 16x16 tile).
// A row-major [M,K] f16, W row-major [N,K] f16 (torch Linear weight layout),
// so both WMMA fragments are contiguous per-lane loads (same k = 16*hi + s order in A and B).
__global__ __launch_bounds__(256) void
k_gemm(const _Float16* __restrict__ A, const _Float16* __restrict__ W,
       const float* __restrict__ bias, float* __restrict__ C,
       int N, int K, int act) {
  const int lane = threadIdx.x & 31;
  const int wave = threadIdx.x >> 5;
  const int m0 = blockIdx.x * 16;
  const int n0 = (blockIdx.y * 8 + wave) * 16;
  if (n0 >= N) return;                       // wave-uniform; EXEC stays all-ones
  const int hi = lane >> 4;                  // 0: lanes 0-15, 1: lanes 16-31
  const _Float16* pA = A + (size_t)(m0 + (lane & 15)) * K + hi * 16;  // A frag: K = kb..kb+15 contiguous
  const _Float16* pB = W + (size_t)(n0 + (lane & 15)) * K + hi * 16;  // B frag: K = kb..kb+15 contiguous
  v8f acc = {};
  for (int k = 0; k < K; k += 32) {
    v8h alo = *(const v8h*)(pA + k);
    v8h ahi = *(const v8h*)(pA + k + 8);
    v16h af;
#pragma unroll
    for (int i = 0; i < 8; ++i) { af[i] = alo[i]; af[i + 8] = ahi[i]; }
    v16h bf = *(const v16h*)(pB + k);
    acc = __builtin_amdgcn_wmma_f32_16x16x32_f16(false, af, false, bf,
                                                 (short)0, acc, false, false);
  }
  const int col = n0 + (lane & 15);
  const float b = bias ? bias[col] : 0.0f;
  float* pC = C + (size_t)(m0 + hi * 8) * N + col;   // C layout: vgpr r -> row m0 + r + 8*hi
#pragma unroll
  for (int r = 0; r < 8; ++r) {
    float v = acc[r] + b;
    if (act == 1) v = tanhf(v);
    pC[(size_t)r * N] = v;
  }
}

// ---------------- GRU gate combine + state update ----------------
// mode 0: plain; mode 1: capture hidden into enc at t==lastIdx[r];
// mode 2: gated update  H = g*h_new + (1-g)*H   (episodic attention)
__global__ void k_gru_update(const float* __restrict__ Gi, const float* __restrict__ Gh,
                             float* __restrict__ Hst, int rows,
                             float* __restrict__ enc, const int* __restrict__ lastIdx,
                             int t, const float* __restrict__ g, int mode) {
  int i = blockIdx.x * blockDim.x + threadIdx.x;
  if (i >= rows * HD) return;
  int r = i / HD, j = i % HD;
  const float* gi = Gi + (size_t)r * G3H;
  const float* gh = Gh + (size_t)r * G3H;
  float rr = 1.0f / (1.0f + expf(-(gi[j] + gh[j])));
  float zz = 1.0f / (1.0f + expf(-(gi[j + HD] + gh[j + HD])));
  float nn = tanhf(gi[j + 2 * HD] + rr * gh[j + 2 * HD]);
  float hp = Hst[i];
  float hn = (1.0f - zz) * nn + zz * hp;
  if (mode == 2) { float gg = g[r]; hn = gg * hn + (1.0f - gg) * hp; }
  Hst[i] = hn;
  if (mode == 1 && t == lastIdx[r]) enc[i] = hn;
}

// ---------------- episodic gate feature  z = [f*q, f*m, |f-q|, |f-m|] -> f16 ----------------
__global__ void k_zfeat(const float* __restrict__ encF, int f,
                        const float* __restrict__ encq, const float* __restrict__ mem,
                        _Float16* __restrict__ z) {
  int i = blockIdx.x * blockDim.x + threadIdx.x;
  if (i >= BD * HD) return;
  int b = i / HD, j = i % HD;
  float fv = encF[((size_t)b * NFD + f) * HD + j];
  float q = encq[i], m = mem[i];
  _Float16* zr = z + (size_t)b * 4 * HD;
  zr[j]          = (_Float16)(fv * q);
  zr[j + HD]     = (_Float16)(fv * m);
  zr[j + 2 * HD] = (_Float16)fabsf(fv - q);
  zr[j + 3 * HD] = (_Float16)fabsf(fv - m);
}

// enc_facts [B,NF,H] f32 -> [NF][B,H] f16 (contiguous per-fact A matrices)
__global__ void k_factT16(const float* __restrict__ encF, _Float16* __restrict__ out) {
  int i = blockIdx.x * blockDim.x + threadIdx.x;
  if (i >= NFD * BD * HD) return;
  int f = i / (BD * HD), rem = i % (BD * HD);
  int b = rem / HD, j = rem % HD;
  out[i] = (_Float16)encF[((size_t)b * NFD + f) * HD + j];
}

// gate MLP second layer: g[b] = sigmoid(dot(t1[b,:], w2) + b2), one wave per row
__global__ void k_gate2(const float* __restrict__ t1, const float* __restrict__ w2,
                        const float* __restrict__ b2, float* __restrict__ g) {
  int wv = (blockIdx.x * blockDim.x + threadIdx.x) >> 5;
  int lane = threadIdx.x & 31;
  if (wv >= BD) return;
  float s = 0.0f;
  for (int j = lane; j < HD; j += 32) s += t1[(size_t)wv * HD + j] * w2[j];
#pragma unroll
  for (int o = 16; o; o >>= 1) s += __shfl_xor(s, o, 32);
  if (lane == 0) g[wv] = 1.0f / (1.0f + expf(-(s + b2[0])));
}

// answer step input: [embed(prev argmax or SEQBEGIN), enc_q] -> f16 [B,512]
__global__ void k_ans_inp(const float* __restrict__ emb, const int* __restrict__ idx,
                          const _Float16* __restrict__ q16, _Float16* __restrict__ inp) {
  int i = blockIdx.x * blockDim.x + threadIdx.x;
  if (i >= BD * 2 * HD) return;
  int b = i / (2 * HD), j = i % (2 * HD);
  if (j < HD) {
    int tk = idx ? idx[b] : SEQB;
    inp[i] = (_Float16)emb[(size_t)tk * HD + j];
  } else {
    inp[i] = q16[(size_t)b * HD + (j - HD)];
  }
}

// log-softmax row of 32000 + argmax (feedback for greedy decode); one block per row
__global__ __launch_bounds__(512) void
k_logsoftmax_argmax(const float* __restrict__ logits, float* __restrict__ out,
                    int step, int* __restrict__ idxOut) {
  __shared__ float smax[512];
  __shared__ int   sidx[512];
  __shared__ float ssum[512];
  int b = blockIdx.x, tid = threadIdx.x;
  const float* row = logits + (size_t)b * VD;
  float m = -INFINITY; int mi = 0;
  for (int v = tid; v < VD; v += blockDim.x) {
    float x = row[v];
    if (x > m) { m = x; mi = v; }
  }
  smax[tid] = m; sidx[tid] = mi;
  __syncthreads();
  for (int s = blockDim.x / 2; s; s >>= 1) {
    if (tid < s) {
      float o = smax[tid + s];
      if (o > smax[tid] || (o == smax[tid] && sidx[tid + s] < sidx[tid])) {
        smax[tid] = o; sidx[tid] = sidx[tid + s];
      }
    }
    __syncthreads();
  }
  float M = smax[0];
  if (tid == 0) idxOut[b] = sidx[0];
  float s = 0.0f;
  for (int v = tid; v < VD; v += blockDim.x) s += expf(row[v] - M);
  ssum[tid] = s;
  __syncthreads();
  for (int k = blockDim.x / 2; k; k >>= 1) {
    if (tid < k) ssum[tid] += ssum[tid + k];
    __syncthreads();
  }
  float lse = M + logf(ssum[0]);
  float* orow = out + ((size_t)b * ALEN + step) * VD;   // [B, alen, V] flattened
  for (int v = tid; v < VD; v += blockDim.x) orow[v] = row[v] - lse;
}

// ---------------------------------------------------------------------------

extern "C" void kernel_launch(void* const* d_in, const int* in_sizes, int n_in,
                              void* d_out, int out_size, void* d_ws, size_t ws_size,
                              hipStream_t stream) {
  const int*   allfacts = (const int*)d_in[0];
  const int*   afmask   = (const int*)d_in[1];
  const int*   questions= (const int*)d_in[2];
  const int*   qmask    = (const int*)d_in[3];
  /* d_in[4] = alen (device scalar, value 8; fixed at compile time) */
  const float* embed_w  = (const float*)d_in[5];
  const float* ig_wih = (const float*)d_in[6],  *ig_whh = (const float*)d_in[7];
  const float* ig_bih = (const float*)d_in[8],  *ig_bhh = (const float*)d_in[9];
  const float* qg_wih = (const float*)d_in[10], *qg_whh = (const float*)d_in[11];
  const float* qg_bih = (const float*)d_in[12], *qg_bhh = (const float*)d_in[13];
  const float* gate_w1= (const float*)d_in[14], *gate_b1= (const float*)d_in[15];
  const float* gate_w2= (const float*)d_in[16], *gate_b2= (const float*)d_in[17];
  const float* att_wih= (const float*)d_in[18], *att_whh= (const float*)d_in[19];
  const float* att_bih= (const float*)d_in[20], *att_bhh= (const float*)d_in[21];
  const float* mem_wih= (const float*)d_in[22], *mem_whh= (const float*)d_in[23];
  const float* mem_bih= (const float*)d_in[24], *mem_bhh= (const float*)d_in[25];
  const float* ans_wih= (const float*)d_in[26], *ans_whh= (const float*)d_in[27];
  const float* ans_bih= (const float*)d_in[28], *ans_bhh= (const float*)d_in[29];
  const float* fc_w   = (const float*)d_in[30], *fc_b   = (const float*)d_in[31];

  const int BN = BD * NFD;   // 1280 fact sequences

  // -------- workspace allocator --------
  char* ws = (char*)d_ws;
  size_t off = 0;
  auto alloc = [&](size_t bytes) -> void* {
    void* p = ws + off;
    off = (off + bytes + 255) & ~(size_t)255;
    return p;
  };
  // f16 weights
  _Float16* w_ig_wih = (_Float16*)alloc((size_t)G3H * HD * 2);
  _Float16* w_ig_whh = (_Float16*)alloc((size_t)G3H * HD * 2);
  _Float16* w_qg_wih = (_Float16*)alloc((size_t)G3H * HD * 2);
  _Float16* w_qg_whh = (_Float16*)alloc((size_t)G3H * HD * 2);
  _Float16* w_att_wih= (_Float16*)alloc((size_t)G3H * HD * 2);
  _Float16* w_att_whh= (_Float16*)alloc((size_t)G3H * HD * 2);
  _Float16* w_mem_wih= (_Float16*)alloc((size_t)G3H * HD * 2);
  _Float16* w_mem_whh= (_Float16*)alloc((size_t)G3H * HD * 2);
  _Float16* w_ans_wih= (_Float16*)alloc((size_t)G3H * 2 * HD * 2);
  _Float16* w_ans_whh= (_Float16*)alloc((size_t)G3H * HD * 2);
  _Float16* w_gate1  = (_Float16*)alloc((size_t)HD * 4 * HD * 2);
  _Float16* w_fc     = (_Float16*)alloc((size_t)VD * HD * 2);
  // state / scratch
  float*    Hf       = (float*)   alloc((size_t)BN * HD * 4);
  _Float16* H16      = (_Float16*)alloc((size_t)BN * HD * 2);
  _Float16* Xt16     = (_Float16*)alloc((size_t)BN * HD * 2);
  float*    Gi       = (float*)   alloc((size_t)BN * G3H * 4);
  float*    Gh       = (float*)   alloc((size_t)BN * G3H * 4);
  int*      lastIdxF = (int*)     alloc((size_t)BN * 4);
  int*      lastIdxQ = (int*)     alloc((size_t)BD * 4);
  float*    encFacts = (float*)   alloc((size_t)BD * NFD * HD * 4);
  _Float16* factT16  = (_Float16*)alloc((size_t)NFD * BD * HD * 2);
  float*    encQ     = (float*)   alloc((size_t)BD * HD * 4);
  _Float16* encQ16   = (_Float16*)alloc((size_t)BD * HD * 2);
  float*    memory   = (float*)   alloc((size_t)BD * HD * 4);
  _Float16* mem16    = (_Float16*)alloc((size_t)BD * HD * 2);
  float*    eSt      = (float*)   alloc((size_t)BD * HD * 4);
  _Float16* e16      = (_Float16*)alloc((size_t)BD * HD * 2);
  float*    hidden   = (float*)   alloc((size_t)BD * HD * 4);
  _Float16* hid16    = (_Float16*)alloc((size_t)BD * HD * 2);
  _Float16* zfeat16  = (_Float16*)alloc((size_t)BD * 4 * HD * 2);
  float*    t1       = (float*)   alloc((size_t)BD * HD * 4);
  float*    gvec     = (float*)   alloc((size_t)BD * 4);
  _Float16* inp16    = (_Float16*)alloc((size_t)BD * 2 * HD * 2);
  float*    logits   = (float*)   alloc((size_t)BD * VD * 4);
  int*      idxbuf   = (int*)     alloc((size_t)BD * 4);

  auto cdiv = [](int a, int b) { return (a + b - 1) / b; };
  auto conv = [&](const float* s, _Float16* d, int n) {
    k_f32_to_f16<<<cdiv(n, 256), 256, 0, stream>>>(s, d, n);
  };
  auto gemm = [&](const _Float16* A, const _Float16* W, const float* bias,
                  float* C, int M, int N, int K, int act) {
    k_gemm<<<dim3(M / 16, N / 128), 256, 0, stream>>>(A, W, bias, C, N, K, act);
  };

  // -------- convert weights to f16 (once per call) --------
  conv(ig_wih,  w_ig_wih,  G3H * HD);  conv(ig_whh,  w_ig_whh,  G3H * HD);
  conv(qg_wih,  w_qg_wih,  G3H * HD);  conv(qg_whh,  w_qg_whh,  G3H * HD);
  conv(att_wih, w_att_wih, G3H * HD);  conv(att_whh, w_att_whh, G3H * HD);
  conv(mem_wih, w_mem_wih, G3H * HD);  conv(mem_whh, w_mem_whh, G3H * HD);
  conv(ans_wih, w_ans_wih, G3H * 2 * HD); conv(ans_whh, w_ans_whh, G3H * HD);
  conv(gate_w1, w_gate1,   HD * 4 * HD);
  conv(fc_w,    w_fc,      VD * HD);

  // -------- fact encoder: GRU over 1280 sequences of 32 tokens --------
  k_last_idx<<<cdiv(BN, 64), 64, 0, stream>>>(afmask, BN, FLD, lastIdxF);
  k_fill_f32<<<cdiv(BN * HD, 256), 256, 0, stream>>>(Hf, 0.0f, BN * HD);
  for (int t = 0; t < FLD; ++t) {
    k_embed_f16<<<cdiv(BN * HD, 256), 256, 0, stream>>>(allfacts, FLD, t, embed_w, Xt16, BN);
    conv(Hf, H16, BN * HD);
    gemm(Xt16, w_ig_wih, ig_bih, Gi, BN, G3H, HD, 0);
    gemm(H16,  w_ig_whh, ig_bhh, Gh, BN, G3H, HD, 0);
    k_gru_update<<<cdiv(BN * HD, 256), 256, 0, stream>>>(Gi, Gh, Hf, BN,
                                                         encFacts, lastIdxF, t, nullptr, 1);
  }

  // -------- question encoder: GRU over 64 sequences of 16 tokens --------
  k_last_idx<<<1, 64, 0, stream>>>(qmask, BD, QLD, lastIdxQ);
  k_fill_f32<<<cdiv(BD * HD, 256), 256, 0, stream>>>(Hf, 0.0f, BD * HD);
  for (int t = 0; t < QLD; ++t) {
    k_embed_f16<<<cdiv(BD * HD, 256), 256, 0, stream>>>(questions, QLD, t, embed_w, Xt16, BD);
    conv(Hf, H16, BD * HD);
    gemm(Xt16, w_qg_wih, qg_bih, Gi, BD, G3H, HD, 0);
    gemm(H16,  w_qg_whh, qg_bhh, Gh, BD, G3H, HD, 0);
    k_gru_update<<<cdiv(BD * HD, 256), 256, 0, stream>>>(Gi, Gh, Hf, BD,
                                                         encQ, lastIdxQ, t, nullptr, 1);
  }
  conv(encQ, encQ16, BD * HD);
  k_factT16<<<cdiv(NFD * BD * HD, 256), 256, 0, stream>>>(encFacts, factT16);

  // -------- episodic memory --------
  k_copy_f32<<<cdiv(BD * HD, 256), 256, 0, stream>>>(encQ, memory, BD * HD);
  for (int ep = 0; ep < 3; ++ep) {
    k_fill_f32<<<cdiv(BD * HD, 256), 256, 0, stream>>>(eSt, 0.0f, BD * HD);
    for (int f = 0; f < NFD; ++f) {
      k_zfeat<<<cdiv(BD * HD, 256), 256, 0, stream>>>(encFacts, f, encQ, memory, zfeat16);
      gemm(zfeat16, w_gate1, gate_b1, t1, BD, HD, 4 * HD, 1);   // tanh(z @ W1^T + b1)
      k_gate2<<<cdiv(BD * 32, 256), 256, 0, stream>>>(t1, gate_w2, gate_b2, gvec);
      conv(eSt, e16, BD * HD);
      gemm(factT16 + (size_t)f * BD * HD, w_att_wih, att_bih, Gi, BD, G3H, HD, 0);
      gemm(e16, w_att_whh, att_bhh, Gh, BD, G3H, HD, 0);
      k_gru_update<<<cdiv(BD * HD, 256), 256, 0, stream>>>(Gi, Gh, eSt, BD,
                                                           nullptr, nullptr, 0, gvec, 2);
    }
    conv(eSt, e16, BD * HD);
    conv(memory, mem16, BD * HD);
    gemm(e16,   w_mem_wih, mem_bih, Gi, BD, G3H, HD, 0);
    gemm(mem16, w_mem_whh, mem_bhh, Gh, BD, G3H, HD, 0);
    k_gru_update<<<cdiv(BD * HD, 256), 256, 0, stream>>>(Gi, Gh, memory, BD,
                                                         nullptr, nullptr, 0, nullptr, 0);
  }

  // -------- answer decoder (greedy feedback, alen = 8) --------
  k_copy_f32<<<cdiv(BD * HD, 256), 256, 0, stream>>>(memory, hidden, BD * HD);
  for (int step = 0; step < ALEN; ++step) {
    k_ans_inp<<<cdiv(BD * 2 * HD, 256), 256, 0, stream>>>(
        embed_w, step ? idxbuf : nullptr, encQ16, inp16);
    conv(hidden, hid16, BD * HD);
    gemm(inp16, w_ans_wih, ans_bih, Gi, BD, G3H, 2 * HD, 0);
    gemm(hid16, w_ans_whh, ans_bhh, Gh, BD, G3H, HD, 0);
    k_gru_update<<<cdiv(BD * HD, 256), 256, 0, stream>>>(Gi, Gh, hidden, BD,
                                                         nullptr, nullptr, 0, nullptr, 0);
    conv(hidden, hid16, BD * HD);
    gemm(hid16, w_fc, fc_b, logits, BD, VD, HD, 0);             // [64,256]@[256,32000]
    k_logsoftmax_argmax<<<BD, 512, 0, stream>>>(logits, (float*)d_out, step, idxbuf);
  }
}
